// OBoW_46772193853837
// MI455X (gfx1250) — matrix operand order repaired and checked by reading, not compiled, not run on hardware
//
#include <hip/hip_runtime.h>
#include <hip/hip_bf16.h>
#include <math.h>

typedef unsigned int u32;
typedef unsigned short u16;
typedef __attribute__((ext_vector_type(16))) __bf16 v16bf;
typedef __attribute__((ext_vector_type(8)))  float  v8f;

// Problem dims
#define BATCH   64
#define HWSZ    144          // 12*12 cropped spatial
#define MTOT    9216         // BATCH*HWSZ
#define NKD     8192         // codebook
#define KD      1024         // channels
#define KD2     512          // channels in packed dwords (2 bf16 per dword)
#define KCH     16           // K-chunk in dwords (32 channels) per stage
#define NCHUNK  (KD2 / KCH)  // 32 chunks

// ---------- f32 -> bf16 split helpers ----------
__device__ __forceinline__ u16 f2bf(float f) {
    u32 u = __float_as_uint(f);
    u32 r = u + 0x7FFFu + ((u >> 16) & 1u);   // round-to-nearest-even
    return (u16)(r >> 16);
}
__device__ __forceinline__ float bf2f(u16 h) { return __uint_as_float(((u32)h) << 16); }

__device__ __forceinline__ void split2(float a, float b, u32& hi, u32& lo) {
    u16 ha = f2bf(a), hb = f2bf(b);
    u16 la = f2bf(a - bf2f(ha)), lb = f2bf(b - bf2f(hb));
    hi = (u32)ha | ((u32)hb << 16);
    lo = (u32)la | ((u32)lb << 16);
}

// ---------- K0: embedding -> bf16 hi/lo + row sumsq ----------
__global__ __launch_bounds__(256) void prep_emb_kernel(
    const float* __restrict__ emb, u32* __restrict__ ehi, u32* __restrict__ elo,
    float* __restrict__ esq)
{
    const int n = blockIdx.x, t = threadIdx.x;
    float4 v = *(const float4*)(emb + (size_t)n * KD + t * 4);
    float ss = v.x * v.x + v.y * v.y + v.z * v.z + v.w * v.w;
    u32 h0, l0, h1, l1;
    split2(v.x, v.y, h0, l0);
    split2(v.z, v.w, h1, l1);
    size_t o = (size_t)n * KD2 + t * 2;
    ehi[o] = h0; ehi[o + 1] = h1;
    elo[o] = l0; elo[o + 1] = l1;
    __shared__ float red[256];
    red[t] = ss; __syncthreads();
    for (int s = 128; s > 0; s >>= 1) { if (t < s) red[t] += red[t + s]; __syncthreads(); }
    if (t == 0) esq[n] = red[0];
}

// ---------- K1: cropped features -> bf16 hi/lo [m][k] + row sumsq ----------
__global__ __launch_bounds__(256) void prep_feat_kernel(
    const float* __restrict__ feat, u32* __restrict__ fhi, u32* __restrict__ flo,
    float* __restrict__ fsq)
{
    const int m = blockIdx.x, t = threadIdx.x;
    const int b = m / HWSZ, hw = m - b * HWSZ;
    const int h = hw / 12 + 1, w = hw % 12 + 1;     // crop [1:13]
    const float* base = feat + (size_t)b * (KD * 196) + h * 14 + w;
    const int c = t * 4;
    float a0 = base[(size_t)(c + 0) * 196];
    float a1 = base[(size_t)(c + 1) * 196];
    float a2 = base[(size_t)(c + 2) * 196];
    float a3 = base[(size_t)(c + 3) * 196];
    float ss = a0 * a0 + a1 * a1 + a2 * a2 + a3 * a3;
    u32 h0, l0, h1, l1;
    split2(a0, a1, h0, l0);
    split2(a2, a3, h1, l1);
    size_t o = (size_t)m * KD2 + t * 2;
    fhi[o] = h0; fhi[o + 1] = h1;
    flo[o] = l0; flo[o + 1] = l1;
    __shared__ float red[256];
    red[t] = ss; __syncthreads();
    for (int s = 128; s > 0; s >>= 1) { if (t < s) red[t] += red[t + s]; __syncthreads(); }
    if (t == 0) fsq[m] = red[0];
}

// ---------- async global -> LDS copy (CDNA5, tracked by ASYNCcnt) ----------
// GVS mode: mem_addr = SADDR(64b) + VADDR(32b offset); LDS dest addr in VDST VGPR.
__device__ __forceinline__ void async_copy_b128(u32 lds_byte_addr, const u32* base, u32 byte_off) {
    asm volatile("global_load_async_to_lds_b128 %0, %1, %2"
                 :: "v"(lds_byte_addr), "v"(byte_off), "s"(base) : "memory");
}
__device__ __forceinline__ void wait_async0() {
    asm volatile("s_wait_asynccnt 0x0" ::: "memory");
}
// generic-pointer low 32 bits == raw LDS byte offset (ISA 10.2 aperture layout)
__device__ __forceinline__ u32 lds_addr32(const void* p) { return (u32)(uintptr_t)p; }

// ---------- WMMA fragment loads from LDS (ISA 7.12.2 layouts), 16-dword rows ----------
// A 16x32 bf16: lane = M%16 (lanes>=16 -> K+8); VGPR j -> dword (j&3) + ((j>>2)<<3)
__device__ __forceinline__ v16bf load_frag_A(const u32* base, int lane) {
    union { v16bf v; u32 u[8]; } f;
    const u32* row = base + (lane & 15) * KCH + ((lane >> 4) << 2);
#pragma unroll
    for (int j = 0; j < 8; ++j)
        f.u[j] = row[(j & 3) + ((j >> 2) << 3)];
    return f.v;
}
// B 32x16 bf16: lane = N%16 (lanes>=16 -> K+8); VGPR j -> dword (j&3) + ((j>>2)<<3)
__device__ __forceinline__ v16bf load_frag_B(const u32* base, int lane) {
    union { v16bf v; u32 u[8]; } f;
    const u32* row = base + (lane & 15) * KCH + ((lane >> 4) << 2);
#pragma unroll
    for (int j = 0; j < 8; ++j)
        f.u[j] = row[(j & 3) + ((j >> 2) << 3)];
    return f.v;
}

// ---------- K2: split-bf16 GEMM -> raw logits, async double-buffered staging ----------
// Block tile: 64 (M) x 128 (N); 8 waves, each 32x32 (2x2 WMMA tiles).
// cross = Ah*Bh + Ah*Bl + Al*Bh + Al*Bl  (near-fp32 via bf16 pairs)
__global__ __launch_bounds__(256) void gemm_logits_kernel(
    const u32* __restrict__ fhi, const u32* __restrict__ flo,
    const u32* __restrict__ ehi, const u32* __restrict__ elo,
    const float* __restrict__ fsq, const float* __restrict__ esq,
    const float* __restrict__ mdm, float* __restrict__ codes)
{
    __shared__ u32 lAh[2][64 * KCH],  lAl[2][64 * KCH];    // 2 x (4KB + 4KB)
    __shared__ u32 lBh[2][128 * KCH], lBl[2][128 * KCH];   // 2 x (8KB + 8KB)
    __shared__ float fsq_s[64];                            // total LDS ~48.3KB

    const int t = threadIdx.x;
    const int m0 = blockIdx.y * 64;
    const int n0 = blockIdx.x * 128;
    if (t < 64) fsq_s[t] = fsq[m0 + t];

    const int lane = t & 31;
    const int wv = t >> 5;
    const int waveM = wv >> 2;   // 0..1  (rows waveM*32 .. +31)
    const int waveN = wv & 3;    // 0..3  (cols waveN*32 .. +31)

    // per-thread staging coordinates (fixed across chunks)
    const int rA = t >> 2, cA = (t & 3) << 2;              // 64 rows x 4 uint4
    const u32 ldsA_h0 = lds_addr32(&lAh[0][rA * KCH + cA]);
    const u32 ldsA_l0 = lds_addr32(&lAl[0][rA * KCH + cA]);
    const u32 ldsA_h1 = lds_addr32(&lAh[1][rA * KCH + cA]);
    const u32 ldsA_l1 = lds_addr32(&lAl[1][rA * KCH + cA]);
    const u32 gAoff   = (u32)(((m0 + rA) * KD2 + cA) * 4);

    v8f acc[2][2];
#pragma unroll
    for (int mi = 0; mi < 2; ++mi)
#pragma unroll
        for (int ni = 0; ni < 2; ++ni)
#pragma unroll
            for (int i = 0; i < 8; ++i) acc[mi][ni][i] = 0.0f;

    // ---- stage one K-chunk (32 channels) into buffer `buf` via async DMA ----
    auto stage = [&](int buf, int kc2) {
        // A hi/lo: one b128 per thread each
        async_copy_b128((buf ? ldsA_h1 : ldsA_h0) , fhi, gAoff + (u32)(kc2 * 4));
        async_copy_b128((buf ? ldsA_l1 : ldsA_l0) , flo, gAoff + (u32)(kc2 * 4));
        // B hi/lo: two b128 per thread each (128 rows x 4 uint4 = 512 items)
#pragma unroll
        for (int i = 0; i < 2; ++i) {
            int item = t + i * 256;
            int r = item >> 2, c4 = (item & 3) << 2;
            u32 go = (u32)(((n0 + r) * KD2 + kc2 + c4) * 4);
            async_copy_b128(lds_addr32(&lBh[buf][r * KCH + c4]), ehi, go);
            async_copy_b128(lds_addr32(&lBl[buf][r * KCH + c4]), elo, go);
        }
    };

    stage(0, 0);                                           // prologue
    for (int c = 0; c < NCHUNK; ++c) {
        const int cur = c & 1;
        wait_async0();          // this wave's copies into buf `cur` landed
        __syncthreads();        // everyone's copies landed; prev buf fully consumed
        if (c + 1 < NCHUNK) stage(cur ^ 1, (c + 1) * KCH); // overlap DMA with math

        v16bf ah[2], al[2], bh[2], bl[2];
#pragma unroll
        for (int mi = 0; mi < 2; ++mi) {
            ah[mi] = load_frag_A(&lAh[cur][(waveM * 32 + mi * 16) * KCH], lane);
            al[mi] = load_frag_A(&lAl[cur][(waveM * 32 + mi * 16) * KCH], lane);
        }
#pragma unroll
        for (int ni = 0; ni < 2; ++ni) {
            bh[ni] = load_frag_B(&lBh[cur][(waveN * 32 + ni * 16) * KCH], lane);
            bl[ni] = load_frag_B(&lBl[cur][(waveN * 32 + ni * 16) * KCH], lane);
        }
#pragma unroll
        for (int mi = 0; mi < 2; ++mi)
#pragma unroll
            for (int ni = 0; ni < 2; ++ni) {
                v8f a = acc[mi][ni];
                a = __builtin_amdgcn_wmma_f32_16x16x32_bf16(false, al[mi], false, bl[ni], (short)0, a, false, false);
                a = __builtin_amdgcn_wmma_f32_16x16x32_bf16(false, al[mi], false, bh[ni], (short)0, a, false, false);
                a = __builtin_amdgcn_wmma_f32_16x16x32_bf16(false, ah[mi], false, bl[ni], (short)0, a, false, false);
                a = __builtin_amdgcn_wmma_f32_16x16x32_bf16(false, ah[mi], false, bh[ni], (short)0, a, false, false);
                acc[mi][ni] = a;
            }
    }

    const float invd = 15.0f / mdm[0];
    // C/D layout: VGPR i, lanes 0-15 -> M=i, N=lane; lanes 16-31 -> M=i+8
#pragma unroll
    for (int mi = 0; mi < 2; ++mi)
#pragma unroll
        for (int ni = 0; ni < 2; ++ni) {
            const int col = n0 + waveN * 32 + ni * 16 + (lane & 15);
            const float esqv = esq[col];
#pragma unroll
            for (int i = 0; i < 8; ++i) {
                const int row = waveM * 32 + mi * 16 + ((lane >> 4) << 3) + i;
                const int m = m0 + row;
                const int b = m / HWSZ, hw = m - b * HWSZ;
                const float logit = invd * (2.0f * acc[mi][ni][i] - fsq_s[row] - esqv);
                codes[((size_t)b * NKD + col) * HWSZ + hw] = logit;
            }
        }
}

// ---------- K3: per-row softmax stats (max, sum of exp) ----------
__global__ __launch_bounds__(256) void softmax_stats_kernel(
    const float* __restrict__ codes, float* __restrict__ rowmax, float* __restrict__ sumexp)
{
    const int m = blockIdx.x, t = threadIdx.x;
    const int b = m / HWSZ, hw = m - b * HWSZ;
    const float* p = codes + (size_t)b * NKD * HWSZ + hw;
    float v[32];
    float mx = -INFINITY;
#pragma unroll
    for (int j = 0; j < 32; ++j) {
        v[j] = p[(size_t)(t + j * 256) * HWSZ];
        mx = fmaxf(mx, v[j]);
    }
    __shared__ float red[256];
    red[t] = mx; __syncthreads();
    for (int s = 128; s > 0; s >>= 1) { if (t < s) red[t] = fmaxf(red[t], red[t + s]); __syncthreads(); }
    const float rmx = red[0]; __syncthreads();
    float sm = 0.0f;
#pragma unroll
    for (int j = 0; j < 32; ++j) sm += __expf(v[j] - rmx);
    red[t] = sm; __syncthreads();
    for (int s = 128; s > 0; s >>= 1) { if (t < s) red[t] += red[t + s]; __syncthreads(); }
    if (t == 0) { rowmax[m] = rmx; sumexp[m] = red[0]; }
}

// ---------- K4: normalize logits -> softmax codes (float4) ----------
__global__ __launch_bounds__(256) void normalize_kernel(
    float* __restrict__ codes, const float* __restrict__ rowmax, const float* __restrict__ sumexp)
{
    const size_t e = ((size_t)blockIdx.x * 256 + threadIdx.x) * 4;
    const size_t tmp = e / HWSZ;
    const int hw = (int)(e - tmp * HWSZ);
    const int b = (int)(tmp / NKD);
    const int mbase = b * HWSZ + hw;          // 4 consecutive hw share (b,k)
    float4 l = *(float4*)(codes + e);
    float4 o;
    o.x = __expf(l.x - rowmax[mbase + 0]) / sumexp[mbase + 0];
    o.y = __expf(l.y - rowmax[mbase + 1]) / sumexp[mbase + 1];
    o.z = __expf(l.z - rowmax[mbase + 2]) / sumexp[mbase + 2];
    o.w = __expf(l.w - rowmax[mbase + 3]) / sumexp[mbase + 3];
    *(float4*)(codes + e) = o;
}

// ---------- K5: bow = L1-normalized max over (h,w) ----------
__global__ __launch_bounds__(1024) void bow_kernel(
    const float* __restrict__ codes, float* __restrict__ bow)
{
    const int b = blockIdx.x, t = threadIdx.x;
    float raw[8];
    float part = 0.0f;
#pragma unroll
    for (int j = 0; j < 8; ++j) {
        const int k = t + j * 1024;
        const float4* p = (const float4*)(codes + ((size_t)b * NKD + k) * HWSZ);
        float mx = 0.0f;                       // softmax outputs are > 0
#pragma unroll 4
        for (int q = 0; q < 36; ++q) {
            float4 v = p[q];
            mx = fmaxf(mx, fmaxf(fmaxf(v.x, v.y), fmaxf(v.z, v.w)));
        }
        raw[j] = mx;
        part += mx;                            // |mx| == mx
    }
    __shared__ float red[1024];
    red[t] = part; __syncthreads();
    for (int s = 512; s > 0; s >>= 1) { if (t < s) red[t] += red[t + s]; __syncthreads(); }
    const float tot = red[0];
#pragma unroll
    for (int j = 0; j < 8; ++j)
        bow[(size_t)b * NKD + t + j * 1024] = raw[j] / tot;
}

extern "C" void kernel_launch(void* const* d_in, const int* in_sizes, int n_in,
                              void* d_out, int out_size, void* d_ws, size_t ws_size,
                              hipStream_t stream) {
    const float* feat = (const float*)d_in[0];   // (64,1024,14,14)
    const float* emb  = (const float*)d_in[1];   // (8192,1024), already clipped >= 0
    const float* mdm  = (const float*)d_in[2];   // (1,)

    float* bow   = (float*)d_out;                        // 64*8192
    float* codes = (float*)d_out + (size_t)BATCH * NKD;  // 64*8192*144

    // Workspace carve-out (~72 MB total)
    char* ws = (char*)d_ws;
    size_t off = 0;
    auto carve = [&](size_t bytes) -> char* {
        char* p = ws + off;
        off = (off + bytes + 255) & ~(size_t)255;
        return p;
    };
    u32*  ehi    = (u32*)carve((size_t)NKD  * KD2 * 4);
    u32*  elo    = (u32*)carve((size_t)NKD  * KD2 * 4);
    u32*  fhi    = (u32*)carve((size_t)MTOT * KD2 * 4);
    u32*  flo    = (u32*)carve((size_t)MTOT * KD2 * 4);
    float* esq   = (float*)carve((size_t)NKD  * 4);
    float* fsq   = (float*)carve((size_t)MTOT * 4);
    float* rmax  = (float*)carve((size_t)MTOT * 4);
    float* sexp  = (float*)carve((size_t)MTOT * 4);

    prep_emb_kernel <<<NKD,  256, 0, stream>>>(emb,  ehi, elo, esq);
    prep_feat_kernel<<<MTOT, 256, 0, stream>>>(feat, fhi, flo, fsq);

    dim3 g2(NKD / 128, MTOT / 64);   // (64, 144)
    gemm_logits_kernel<<<g2, 256, 0, stream>>>(fhi, flo, ehi, elo, fsq, esq, mdm, codes);

    softmax_stats_kernel<<<MTOT, 256, 0, stream>>>(codes, rmax, sexp);

    const size_t nelem = (size_t)BATCH * NKD * HWSZ;     // 75,497,472 (divisible by 1024)
    normalize_kernel<<<(unsigned)(nelem / 1024), 256, 0, stream>>>(codes, rmax, sexp);

    bow_kernel<<<BATCH, 1024, 0, stream>>>(codes, bow);
}
